// GCN_13692355740362
// MI455X (gfx1250) — compile-verified, matrix operand order checked by reading, not run on hardware
//
#include <hip/hip_runtime.h>

// ---------------------------------------------------------------------------
// Types for CDNA5 WMMA / TDM (gfx1250, wave32)
// ---------------------------------------------------------------------------
typedef __attribute__((ext_vector_type(16))) __bf16 v16bf;
typedef __attribute__((ext_vector_type(8)))  float  v8f;
typedef __attribute__((ext_vector_type(4))) unsigned int u32x4;
typedef __attribute__((ext_vector_type(8))) int i32x8;
typedef __attribute__((ext_vector_type(4))) int i32x4;

union FragBF {
  v16bf v;
  uint4 q[2];   // 32 bytes = 16 bf16
};

__device__ __forceinline__ unsigned short f32_to_bf16(float f) {
  unsigned u = __float_as_uint(f);
  unsigned r = 0x7FFFu + ((u >> 16) & 1u);   // round-to-nearest-even
  return (unsigned short)((u + r) >> 16);
}

// ---------------------------------------------------------------------------
// TDM: issue a 2D tile load (tile_k x tile_rows, bf16) into LDS.
// Descriptor packing per CDNA5 ISA 8.3-8.5:
//   group0: count=1 | lds_addr | global_addr(57b) | type=2
//   group1: data_size=1(2B), pad_enable, pad_interval=3 (64B), pad_amount=3
//           (4 DWORDs = 16B)  -> LDS row stride 80B = 40 elements,
//           tensor_dim0=K, tensor_dim1=rows remaining (OOB rows read 0),
//           tile_dim0=KC, tile_dim1=tile_rows, dim0 stride=K elements
// This toolchain exposes the 6-arg builtin:
//   (u32x4 g0, i32x8 g1, i32x4 g2, i32x4 g3, i32x8 extra, i32 cpol)
// ---------------------------------------------------------------------------
union TdmG0 { u32x4 v; unsigned w[4]; };
union TdmG1 { i32x8 v; int w[8]; };
union TdmG2 { i32x4 v; int w[4]; };

__device__ __forceinline__ void tdm_load_tile_2d(unsigned lds_off,
                                                 const unsigned short* gptr,
                                                 int tile_k, int tile_rows,
                                                 int tensor_k,
                                                 long long rows_remain,
                                                 int stride_elems) {
  TdmG0 g0 = {};
  unsigned long long ga = (unsigned long long)(uintptr_t)gptr;
  g0.w[0] = 1u;                                    // count=1 (valid descriptor)
  g0.w[1] = lds_off;                               // lds_addr
  g0.w[2] = (unsigned)ga;                          // global_addr[31:0]
  g0.w[3] = (unsigned)((ga >> 32) & 0x1FFFFFFull)  // global_addr[56:32]
            | (2u << 30);                          // type=2 (image)
  TdmG1 g1 = {};
  unsigned td0 = (unsigned)tensor_k;
  unsigned td1 = rows_remain < 0 ? 0u : (unsigned)rows_remain;
  g1.w[0] = (1 << 16)          // data_size = 1 -> 2 bytes
          | (1 << 20)          // pad_enable
          | (3 << 22)          // pad_interval: 16 DWORDs (64B) before pad
          | (3 << 25);         // pad_amount: 4 DWORDs (16B)
  g1.w[1] = (int)((td0 & 0xFFFFu) << 16);                          // tensor_dim0 lo
  g1.w[2] = (int)(((td0 >> 16) & 0xFFFFu) | ((td1 & 0xFFFFu) << 16));
  g1.w[3] = (int)(((td1 >> 16) & 0xFFFFu) |
                  (((unsigned)tile_k & 0xFFFFu) << 16));           // tile_dim0
  g1.w[4] = (int)((unsigned)tile_rows & 0xFFFFu);                  // tile_dim1
  g1.w[5] = stride_elems;                                          // dim0 stride lo
  g1.w[6] = 0;
  g1.w[7] = 0;
  TdmG2 g2 = {};
  TdmG2 g3 = {};
  TdmG1 gx = {};   // extra group (zero-filled; 2D tensor uses groups 0-1 only)
  __builtin_amdgcn_tensor_load_to_lds(g0.v, g1.v, g2.v, g3.v, gx.v, 0);
}

// ---------------------------------------------------------------------------
// Conversion kernels
// ---------------------------------------------------------------------------
__global__ void k_convert_bf16(const float* __restrict__ src,
                               unsigned short* __restrict__ dst, size_t n) {
  size_t i = (size_t)blockIdx.x * blockDim.x + threadIdx.x;
  if (i < n) dst[i] = f32_to_bf16(src[i]);
}

// W: K x M (row-major) fp32 -> Wt: M x K (row-major) bf16
__global__ void k_convert_transpose(const float* __restrict__ W,
                                    unsigned short* __restrict__ Wt,
                                    int K, int M) {
  size_t i = (size_t)blockIdx.x * blockDim.x + threadIdx.x;
  if (i >= (size_t)K * M) return;
  int k = (int)(i / M);
  int m = (int)(i % M);
  Wt[(size_t)m * K + k] = f32_to_bf16(W[i]);
}

// ---------------------------------------------------------------------------
// Tiled bf16 WMMA GEMM with TDM double-buffered staging:
//   C[N x M] = A[N x K] * B[K x M];  B supplied transposed (Bt: M x K bf16)
//   block = 256 threads = 8 waves; block tile 128 x BN
//   wave (wr,wc): 32 rows x BN/2 cols = 2 x (BN/32) tiles of
//   v_wmma_f32_16x16x32_bf16.  Wave 0 drives the Tensor Data Mover.
// ---------------------------------------------------------------------------
#define BM 128
#define KC 32
#define LD 40   // LDS row stride in elements (32 + 8 pad) == TDM pad fields

template <int BN>
__global__ __launch_bounds__(256)
void k_gemm_bf16_wmma_tdm(const unsigned short* __restrict__ A,
                          const unsigned short* __restrict__ Bt,
                          float* __restrict__ C,
                          int Nrows, int K, int M) {
  constexpr int TI = 2;          // 16-row tiles per wave
  constexpr int TJ = BN / 32;    // 16-col tiles per wave
  __shared__ __align__(16) unsigned short lA[2][BM * LD];
  __shared__ __align__(16) unsigned short lB[2][BN * LD];

  const int tid  = threadIdx.x;
  const int lane = tid & 31;
  const int wave = tid >> 5;      // 0..7
  const int wr   = wave & 3;      // row group of 32
  const int wc   = wave >> 2;     // col group of BN/2 (0..1)

  const int rowBase = blockIdx.x * BM;
  const int colBase = blockIdx.y * BN;

  v8f acc[TI][TJ];
#pragma unroll
  for (int i = 0; i < TI; ++i)
#pragma unroll
    for (int j = 0; j < TJ; ++j) acc[i][j] = v8f{};

  // 16-bit WMMA A/B fragment layout (ISA 7.12.2):
  // lanes 0-15: M/N = lane, K in {0..7, 16..23}; lanes 16-31: K in {8..15, 24..31}
  const int lrow = lane & 15;
  const int koff = (lane >> 4) << 3;   // 0 or 8 elements

  // prologue: stage tile 0 into buffer 0
  if (wave == 0) {
    tdm_load_tile_2d((unsigned)(uintptr_t)&lA[0][0], A + (size_t)rowBase * K,
                     KC, BM, K, (long long)Nrows - rowBase, K);
    tdm_load_tile_2d((unsigned)(uintptr_t)&lB[0][0], Bt + (size_t)colBase * K,
                     KC, BN, K, (long long)M - colBase, K);
  }

  int s = 0;
  for (int k0 = 0; k0 < K; k0 += KC, s ^= 1) {
    const bool more = (k0 + KC) < K;
    if (wave == 0) {
      if (more) {
        // prefetch next tile into the other buffer, then wait for current
        tdm_load_tile_2d((unsigned)(uintptr_t)&lA[s ^ 1][0],
                         A + (size_t)rowBase * K + (k0 + KC),
                         KC, BM, K, (long long)Nrows - rowBase, K);
        tdm_load_tile_2d((unsigned)(uintptr_t)&lB[s ^ 1][0],
                         Bt + (size_t)colBase * K + (k0 + KC),
                         KC, BN, K, (long long)M - colBase, K);
        __builtin_amdgcn_s_wait_tensorcnt(2);   // tile k complete, k+1 in flight
      } else {
        __builtin_amdgcn_s_wait_tensorcnt(0);
      }
    }
    __syncthreads();   // tile s visible to all waves

    FragBF a[TI], b[TJ];
#pragma unroll
    for (int i = 0; i < TI; ++i) {
      int r = wr * 32 + i * 16 + lrow;
      a[i].q[0] = *(const uint4*)(&lA[s][r * LD + koff]);
      a[i].q[1] = *(const uint4*)(&lA[s][r * LD + koff + 16]);
    }
#pragma unroll
    for (int j = 0; j < TJ; ++j) {
      int n = wc * (BN / 2) + j * 16 + lrow;
      b[j].q[0] = *(const uint4*)(&lB[s][n * LD + koff]);
      b[j].q[1] = *(const uint4*)(&lB[s][n * LD + koff + 16]);
    }

#pragma unroll
    for (int i = 0; i < TI; ++i)
#pragma unroll
      for (int j = 0; j < TJ; ++j)
        acc[i][j] = __builtin_amdgcn_wmma_f32_16x16x32_bf16(
            false, a[i].v, false, b[j].v, (short)0, acc[i][j], false, false);

    __syncthreads();   // all waves done reading buf s before it is re-filled
  }

  // store: C/D layout: vgpr r -> row (lane<16 ? r : r+8), col = lane&15
  const int mo = (lane < 16) ? 0 : 8;
  const int nn = lane & 15;
#pragma unroll
  for (int i = 0; i < TI; ++i)
#pragma unroll
    for (int j = 0; j < TJ; ++j)
#pragma unroll
      for (int r = 0; r < 8; ++r) {
        int grow = rowBase + wr * 32 + i * 16 + mo + r;
        int gcol = colBase + wc * (BN / 2) + j * 16 + nn;
        if (grow < Nrows)
          C[(size_t)grow * M + gcol] = acc[i][j][r];
      }
}

// ---------------------------------------------------------------------------
// SpMM scatter:  out[row[e], :] += val[e] * support[col[e], :]
// one wave32 per edge
// ---------------------------------------------------------------------------
__global__ __launch_bounds__(256)
void k_spmm_scatter(const int* __restrict__ rows, const int* __restrict__ cols,
                    const float* __restrict__ vals,
                    const float* __restrict__ support,
                    float* __restrict__ out, int E, int W) {
  int e = blockIdx.x * (blockDim.x >> 5) + (threadIdx.x >> 5);
  if (e >= E) return;
  int lane = threadIdx.x & 31;
  int r = rows[e], c = cols[e];
  float v = vals[e];
  const float* src = support + (size_t)c * W;
  float* dst = out + (size_t)r * W;
  for (int f = lane; f < W; f += 32)
    atomicAdd(&dst[f], v * src[f]);
}

// ---------------------------------------------------------------------------
// Fused elementwise kernels
// ---------------------------------------------------------------------------
__global__ void k_fuse_proj(const float* __restrict__ agg,
                            const float* __restrict__ res,
                            const float* __restrict__ b,
                            const float* __restrict__ pb,
                            float* __restrict__ hf,
                            unsigned short* __restrict__ hb,
                            size_t total, int M) {
  size_t i = (size_t)blockIdx.x * blockDim.x + threadIdx.x;
  if (i >= total) return;
  int j = (int)(i % M);
  float v = fmaxf(agg[i] + b[j] + res[i] + pb[j], 0.0f);
  hf[i] = v;
  hb[i] = f32_to_bf16(v);
}

__global__ void k_fuse_res(const float* __restrict__ agg,
                           const float* __restrict__ b,
                           float* __restrict__ hf,
                           unsigned short* __restrict__ hb,
                           size_t total, int M) {
  size_t i = (size_t)blockIdx.x * blockDim.x + threadIdx.x;
  if (i >= total) return;
  int j = (int)(i % M);
  float v = fmaxf(agg[i] + b[j] + hf[i], 0.0f);
  hf[i] = v;
  hb[i] = f32_to_bf16(v);
}

// ---------------------------------------------------------------------------
// Final: logits = relu(agg + b2 + res + p2b); out = log_softmax(logits)
// NCLASS = 64, one wave32 per node row, 2 values per lane
// ---------------------------------------------------------------------------
__global__ __launch_bounds__(256)
void k_final_logsoftmax(const float* __restrict__ agg,
                        const float* __restrict__ res,
                        const float* __restrict__ b2,
                        const float* __restrict__ p2b,
                        float* __restrict__ out, int Nrows) {
  int rowid = blockIdx.x * (blockDim.x >> 5) + (threadIdx.x >> 5);
  if (rowid >= Nrows) return;
  int lane = threadIdx.x & 31;
  const int C = 64;
  size_t base = (size_t)rowid * C;
  float z0 = fmaxf(agg[base + lane]      + b2[lane]      + res[base + lane]      + p2b[lane],      0.0f);
  float z1 = fmaxf(agg[base + lane + 32] + b2[lane + 32] + res[base + lane + 32] + p2b[lane + 32], 0.0f);
  float m = fmaxf(z0, z1);
  for (int off = 16; off; off >>= 1) m = fmaxf(m, __shfl_xor(m, off, 32));
  float s = __expf(z0 - m) + __expf(z1 - m);
  for (int off = 16; off; off >>= 1) s += __shfl_xor(s, off, 32);
  float lse = m + __logf(s);
  out[base + lane]      = z0 - lse;
  out[base + lane + 32] = z1 - lse;
}

// ---------------------------------------------------------------------------
// Launch orchestration
// ---------------------------------------------------------------------------
static inline int cdiv(int a, int b) { return (a + b - 1) / b; }
static inline size_t cdivz(size_t a, size_t b) { return (a + b - 1) / b; }

extern "C" void kernel_launch(void* const* d_in, const int* in_sizes, int n_in,
                              void* d_out, int out_size, void* d_ws, size_t ws_size,
                              hipStream_t stream) {
  const float* x       = (const float*)d_in[0];
  const int*   adj_row = (const int*)  d_in[1];
  const int*   adj_col = (const int*)  d_in[2];
  const float* adj_val = (const float*)d_in[3];
  const float* W1  = (const float*)d_in[4];
  const float* b1  = (const float*)d_in[5];
  const float* p1W = (const float*)d_in[6];
  const float* p1b = (const float*)d_in[7];
  const float* W3  = (const float*)d_in[8];
  const float* b3  = (const float*)d_in[9];
  const float* W2  = (const float*)d_in[10];
  const float* b2  = (const float*)d_in[11];
  const float* p2W = (const float*)d_in[12];
  const float* p2b = (const float*)d_in[13];

  const int NFEAT = 1024, NHID = 512, NCLASS = 64;
  const int N = in_sizes[0] / NFEAT;
  const int E = in_sizes[1];

  // ---- carve workspace (256B aligned) ----
  char* p = (char*)d_ws;
  auto carve = [&](size_t bytes) -> void* {
    void* r = (void*)p;
    p += (bytes + 255) & ~(size_t)255;
    return r;
  };
  unsigned short* xb   = (unsigned short*)carve((size_t)N * NFEAT * 2);
  unsigned short* hb   = (unsigned short*)carve((size_t)N * NHID * 2);
  unsigned short* W1t  = (unsigned short*)carve((size_t)NFEAT * NHID * 2);
  unsigned short* p1Wt = (unsigned short*)carve((size_t)NFEAT * NHID * 2);
  unsigned short* W3t  = (unsigned short*)carve((size_t)NHID * NHID * 2);
  unsigned short* W2t  = (unsigned short*)carve((size_t)NHID * NCLASS * 2);
  unsigned short* p2Wt = (unsigned short*)carve((size_t)NHID * NCLASS * 2);
  float* sbuf = (float*)carve((size_t)N * NHID * 4);
  float* rbuf = (float*)carve((size_t)N * NHID * 4);
  float* agg  = (float*)carve((size_t)N * NHID * 4);
  float* hf   = (float*)carve((size_t)N * NHID * 4);

  const int TPB = 256;

  // ---- conversions ----
  {
    size_t n = (size_t)N * NFEAT;
    k_convert_bf16<<<(unsigned)cdivz(n, TPB), TPB, 0, stream>>>(x, xb, n);
  }
  k_convert_transpose<<<(unsigned)cdivz((size_t)NFEAT * NHID, TPB), TPB, 0, stream>>>(W1,  W1t,  NFEAT, NHID);
  k_convert_transpose<<<(unsigned)cdivz((size_t)NFEAT * NHID, TPB), TPB, 0, stream>>>(p1W, p1Wt, NFEAT, NHID);
  k_convert_transpose<<<(unsigned)cdivz((size_t)NHID * NHID,  TPB), TPB, 0, stream>>>(W3,  W3t,  NHID, NHID);
  k_convert_transpose<<<(unsigned)cdivz((size_t)NHID * NCLASS,TPB), TPB, 0, stream>>>(W2,  W2t,  NHID, NCLASS);
  k_convert_transpose<<<(unsigned)cdivz((size_t)NHID * NCLASS,TPB), TPB, 0, stream>>>(p2W, p2Wt, NHID, NCLASS);

  dim3 gemm_grid_h(cdiv(N, BM), NHID / 128);
  const int spmm_blocks = cdiv(E, TPB / 32);
  const size_t totalH = (size_t)N * NHID;

  // ---- layer 1: gc_proj (1024 -> 512) ----
  k_gemm_bf16_wmma_tdm<128><<<gemm_grid_h, TPB, 0, stream>>>(xb, W1t,  sbuf, N, NFEAT, NHID);
  k_gemm_bf16_wmma_tdm<128><<<gemm_grid_h, TPB, 0, stream>>>(xb, p1Wt, rbuf, N, NFEAT, NHID);
  (void)hipMemsetAsync(agg, 0, totalH * 4, stream);
  k_spmm_scatter<<<spmm_blocks, TPB, 0, stream>>>(adj_row, adj_col, adj_val, sbuf, agg, E, NHID);
  k_fuse_proj<<<(unsigned)cdivz(totalH, TPB), TPB, 0, stream>>>(agg, rbuf, b1, p1b, hf, hb, totalH, NHID);

  // ---- layers 2..5: gc_res with shared W3 (512 -> 512) ----
  for (int it = 0; it < 4; ++it) {
    k_gemm_bf16_wmma_tdm<128><<<gemm_grid_h, TPB, 0, stream>>>(hb, W3t, sbuf, N, NHID, NHID);
    (void)hipMemsetAsync(agg, 0, totalH * 4, stream);
    k_spmm_scatter<<<spmm_blocks, TPB, 0, stream>>>(adj_row, adj_col, adj_val, sbuf, agg, E, NHID);
    k_fuse_res<<<(unsigned)cdivz(totalH, TPB), TPB, 0, stream>>>(agg, b3, hf, hb, totalH, NHID);
  }

  // ---- layer 6: gc_proj (512 -> 64) + log_softmax ----
  dim3 gemm_grid_c(cdiv(N, BM), 1);
  const size_t totalC = (size_t)N * NCLASS;
  k_gemm_bf16_wmma_tdm<64><<<gemm_grid_c, TPB, 0, stream>>>(hb, W2t,  sbuf, N, NHID, NCLASS);
  k_gemm_bf16_wmma_tdm<64><<<gemm_grid_c, TPB, 0, stream>>>(hb, p2Wt, rbuf, N, NHID, NCLASS);
  (void)hipMemsetAsync(agg, 0, totalC * 4, stream);
  k_spmm_scatter<<<spmm_blocks, TPB, 0, stream>>>(adj_row, adj_col, adj_val, sbuf, agg, E, NCLASS);
  k_final_logsoftmax<<<cdiv(N, TPB / 32), TPB, 0, stream>>>(agg, rbuf, b2, p2b, (float*)d_out, N);
}
